// TimeSpaceAttention_32684701123117
// MI455X (gfx1250) — compile-verified, matrix operand order checked by reading, not run on hardware
//
#include <hip/hip_runtime.h>
#include <hip/hip_bf16.h>

// ---------------------------------------------------------------------------
// Divided space-time attention for MI455X (gfx1250), bf16 WMMA pipeline.
//   E=768 H=12 d=64 T=8 S=196 B=16  -> M = L*B = 25088 rows
// ---------------------------------------------------------------------------

typedef __attribute__((ext_vector_type(8)))  __bf16 v8bf;
typedef __attribute__((ext_vector_type(16))) __bf16 v16bf;
typedef __attribute__((ext_vector_type(8)))  float  v8f;
typedef __attribute__((ext_vector_type(4)))  unsigned int u32x4;
typedef __attribute__((ext_vector_type(8)))  int i32x8;
typedef __attribute__((ext_vector_type(4)))  int i32x4;

__device__ __forceinline__ unsigned short f32_to_bf16_bits(float f) {
    return __builtin_bit_cast(unsigned short, (__bf16)f);   // v_cvt_bf16_f32
}
__device__ __forceinline__ float bf16_bits_to_f32(unsigned short h) {
    return (float)__builtin_bit_cast(__bf16, h);
}
__device__ __forceinline__ v8f v8f_zero() {
    v8f z;
#pragma unroll
    for (int i = 0; i < 8; ++i) z[i] = 0.0f;
    return z;
}
__device__ __forceinline__ v8f wmma_bf16(v16bf a, v16bf b, v8f c) {
    // D = A(16x32 bf16) * B(32x16 bf16) + C(16x16 f32)
    return __builtin_amdgcn_wmma_f32_16x16x32_bf16(
        false, a, false, b, (short)0, c, false, false);
}

// Load one 16x32 bf16 operand fragment for this lane from a row-major tile.
// A-operand: rowBase = row m of this lane.  B-operand (kept N x K row-major):
// rowBase = row n of this lane.  Lane halves (lane>=16) hold K {8..15,24..31};
// low halves hold K {0..7,16..23} (mirrors ISA A layout).
__device__ __forceinline__ v16bf load_frag_row(const unsigned short* rowBase, int halfSel) {
    const v8bf* p0 = (const v8bf*)(rowBase + halfSel * 8);
    const v8bf* p1 = (const v8bf*)(rowBase + halfSel * 8 + 16);
    v8bf lo = *p0, hi = *p1;
    return __builtin_shufflevector(lo, hi, 0,1,2,3,4,5,6,7,8,9,10,11,12,13,14,15);
}

// ---------------------------------------------------------------------------
// TDM: 2D tensor_load_to_lds of a [rows x 64]-bf16 tile with hardware row
// padding to LDS stride 72 bf16 (pad_interval code 4 = 32 DWORDs = one row,
// pad_amount code 3 = 4 DWORDs = 8 bf16).  Issue from ONE wave only.
// ---------------------------------------------------------------------------
__device__ __forceinline__ void tdm_load_rows64_pad72(const unsigned short* gsrc,
                                                      unsigned ldsByteAddr,
                                                      unsigned rows) {
    const unsigned long long ga = (unsigned long long)(uintptr_t)gsrc;
    u32x4 g0;
    g0[0] = 1u;                                   // count=1, user descriptor
    g0[1] = ldsByteAddr;                          // lds_addr
    g0[2] = (unsigned)ga;                         // global_addr[31:0]
    g0[3] = (unsigned)((ga >> 32) & 0x01ffffffull) | 0x80000000u; // addr[56:32], type=2

    i32x8 g1;
    g1[0] = (int)((1u << 16) | (1u << 20) | (4u << 22) | (3u << 25));
    //             data_size=2B  pad_en      interval=32dw  pad=4dw
    g1[1] = (int)(64u << 16);                     // tensor_dim0 = 64
    g1[2] = (int)(rows << 16);                    // tensor_dim1 = rows
    g1[3] = (int)(64u << 16);                     // tile_dim0 = 64
    g1[4] = (int)rows;                            // tile_dim1 = rows
    g1[5] = 64;                                   // tensor_dim0_stride = 64
    g1[6] = 0;
    g1[7] = 0;

    i32x4 gz4 = {0, 0, 0, 0};
#if defined(__clang_major__) && (__clang_major__ >= 23)
    i32x8 gz8 = {0, 0, 0, 0, 0, 0, 0, 0};
    __builtin_amdgcn_tensor_load_to_lds(g0, g1, gz4, gz4, gz8, 0);
#else
    __builtin_amdgcn_tensor_load_to_lds(g0, g1, gz4, gz4, 0);
#endif
    __builtin_amdgcn_s_wait_tensorcnt((short)0);
}

// ---------------------------------------------------------------------------
// f32 -> bf16 bulk convert
// ---------------------------------------------------------------------------
__global__ void tsa_convert_bf16(const float* __restrict__ src,
                                 unsigned short* __restrict__ dst, int n) {
    int i = blockIdx.x * blockDim.x + threadIdx.x;
    int stride = gridDim.x * blockDim.x;
    for (; i < n; i += stride) dst[i] = f32_to_bf16_bits(src[i]);
}

// ---------------------------------------------------------------------------
// GEMM: C[M,N] = A[M,K](bf16) * Bw[N,K]^T(bf16) + bias[N]
// Block tile 128x128, 8 waves (2x4) of 64x32, K-step 32 (one WMMA K).
// M,N multiples of 128; K multiple of 32 (exact here: no remainders).
// ---------------------------------------------------------------------------
template<int STORE_F32>
__global__ __launch_bounds__(256)
void tsa_gemm_bf16(const unsigned short* __restrict__ A,
                   const unsigned short* __restrict__ Bw,
                   const float* __restrict__ bias,
                   float* __restrict__ Cf,
                   unsigned short* __restrict__ Cb,
                   int M, int N, int K) {
    __shared__ unsigned short Al[128 * 40];   // stride 40 bf16: 16B aligned, conflict-free
    __shared__ unsigned short Bl[128 * 40];

    const int tid  = threadIdx.x;
    const int wave = tid >> 5, lane = tid & 31;
    const int halfSel = lane >> 4, lrow = lane & 15;
    const int waveM = wave >> 2, waveN = wave & 3;
    const int mBase = blockIdx.y * 128;
    const int nBase = blockIdx.x * 128;

    v8f acc[4][2];
#pragma unroll
    for (int t = 0; t < 4; ++t)
#pragma unroll
        for (int u = 0; u < 2; ++u) acc[t][u] = v8f_zero();

    const int ldRow = tid >> 1;
    const int ldSeg = (tid & 1) * 16;
    const unsigned short* gA = A + (size_t)(mBase + ldRow) * K + ldSeg;
    const unsigned short* gB = Bw + (size_t)(nBase + ldRow) * K + ldSeg;

    for (int kb = 0; kb < K; kb += 32) {
        // cooperative staging: 16 bf16 per thread per tile (2 x b128)
        {
            const uint4* ga = (const uint4*)(gA + kb);
            uint4* da = (uint4*)(Al + ldRow * 40 + ldSeg);
            da[0] = ga[0]; da[1] = ga[1];
            const uint4* gb = (const uint4*)(gB + kb);
            uint4* db = (uint4*)(Bl + ldRow * 40 + ldSeg);
            db[0] = gb[0]; db[1] = gb[1];
        }
        if (kb + 32 < K) {       // prefetch next K-step (global_prefetch_b8)
            __builtin_prefetch(gA + kb + 32, 0, 1);
            __builtin_prefetch(gB + kb + 32, 0, 1);
        }
        __syncthreads();

        v16bf afrag[4], bfrag[2];
#pragma unroll
        for (int t = 0; t < 4; ++t)
            afrag[t] = load_frag_row(Al + (waveM * 64 + t * 16 + lrow) * 40, halfSel);
#pragma unroll
        for (int u = 0; u < 2; ++u)
            bfrag[u] = load_frag_row(Bl + (waveN * 32 + u * 16 + lrow) * 40, halfSel);
#pragma unroll
        for (int t = 0; t < 4; ++t)
#pragma unroll
            for (int u = 0; u < 2; ++u)
                acc[t][u] = wmma_bf16(afrag[t], bfrag[u], acc[t][u]);
        __syncthreads();
    }

    // epilogue: C layout VGPR j -> row m = j + 8*half, lane 0..15 -> col n
#pragma unroll
    for (int t = 0; t < 4; ++t)
#pragma unroll
        for (int u = 0; u < 2; ++u) {
            const int col = nBase + waveN * 32 + u * 16 + lrow;
            const float bv = bias[col];
#pragma unroll
            for (int j = 0; j < 8; ++j) {
                const int row = mBase + waveM * 64 + t * 16 + halfSel * 8 + j;
                const float v = acc[t][u][j] + bv;
                if (STORE_F32) Cf[(size_t)row * N + col] = v;
                else           Cb[(size_t)row * N + col] = f32_to_bf16_bits(v);
            }
        }
}

// ---------------------------------------------------------------------------
// Temporal attention (t=8 per spatial site) — tiny, VALU + shuffles.
// One wave per (bh, si). lane = tq*4 + c; lane owns 16 of the 64 d-elems.
// QKV is bf16 [25088 x 2304]; row r = (si*8+t)*16 + b; q/k/v col base 0/768/1536.
// Output XT bf16 [bh][t][196][64].
// ---------------------------------------------------------------------------
__global__ __launch_bounds__(256)
void tsa_temporal_attn(const unsigned short* __restrict__ QKV,
                       unsigned short* __restrict__ XT) {
    const int tid  = threadIdx.x;
    const int wave = tid >> 5, lane = tid & 31;
    const int task = blockIdx.x * 8 + wave;
    if (task >= 192 * 196) return;
    const int bh = task / 196, si = task % 196;
    const int b = bh / 12, h = bh % 12;
    const int tq = lane >> 2;
    const int dOff = (lane & 3) * 16;
    const int colQ = h * 64 + dOff;

    float qv[16];
    {
        const unsigned short* p = QKV + (size_t)((si * 8 + tq) * 16 + b) * 2304 + colQ;
#pragma unroll
        for (int i = 0; i < 16; ++i) qv[i] = bf16_bits_to_f32(p[i]);
    }
    float logit[8];
#pragma unroll
    for (int tk = 0; tk < 8; ++tk) {
        const unsigned short* p =
            QKV + (size_t)((si * 8 + tk) * 16 + b) * 2304 + 768 + colQ;
        float s = 0.f;
#pragma unroll
        for (int i = 0; i < 16; ++i) s += qv[i] * bf16_bits_to_f32(p[i]);
        s += __shfl_xor(s, 1, 32);
        s += __shfl_xor(s, 2, 32);
        logit[tk] = s * 0.125f;                  // d^-0.5 = 64^-0.5
    }
    float mx = logit[0];
#pragma unroll
    for (int tk = 1; tk < 8; ++tk) mx = fmaxf(mx, logit[tk]);
    float sum = 0.f, pr[8];
#pragma unroll
    for (int tk = 0; tk < 8; ++tk) { pr[tk] = __expf(logit[tk] - mx); sum += pr[tk]; }
    const float inv = 1.0f / sum;

    float ov[16];
#pragma unroll
    for (int i = 0; i < 16; ++i) ov[i] = 0.f;
#pragma unroll
    for (int tk = 0; tk < 8; ++tk) {
        const unsigned short* p =
            QKV + (size_t)((si * 8 + tk) * 16 + b) * 2304 + 1536 + colQ;
        const float w = pr[tk] * inv;
#pragma unroll
        for (int i = 0; i < 16; ++i) ov[i] += w * bf16_bits_to_f32(p[i]);
    }
    unsigned short* o = XT + (size_t)((bh * 8 + tq) * 196 + si) * 64 + dOff;
#pragma unroll
    for (int i = 0; i < 16; ++i) o[i] = f32_to_bf16_bits(ov[i]);
}

// ---------------------------------------------------------------------------
// Spatial attention: one workgroup per (bh, ti).  x = softmax(A A^T * scale) A
// A is 196x64 (padded to 208 rows), staged by the Tensor Data Mover with
// hardware row padding.  All GEMMs via WMMA through LDS.
// Output XO bf16 [25088 x 768]: row (si*8+ti)*16 + b, col h*64 + dd.
// ---------------------------------------------------------------------------
__global__ __launch_bounds__(256)
void tsa_spatial_attn(const unsigned short* __restrict__ XT,
                      unsigned short* __restrict__ XO) {
    __shared__ unsigned short Asp[208 * 72];   // A  row-major, stride 72
    __shared__ unsigned short Atp[64 * 232];   // A^T row-major, stride 232
    __shared__ unsigned short Sp [208 * 232];  // scores/probs, K-padded to 224

    const int tid  = threadIdx.x;
    const int wave = tid >> 5, lane = tid & 31;
    const int halfSel = lane >> 4, lrow = lane & 15;
    const int bh = blockIdx.x >> 3, ti = blockIdx.x & 7;
    const int b = bh / 12, h = bh % 12;

    for (int i = tid; i < 208 * 72; i += 256) Asp[i] = 0;
    for (int i = tid; i < 64 * 232; i += 256) Atp[i] = 0;
    __syncthreads();

    // ---- TDM: stage A tile (196 rows x 64 bf16) into Asp with stride 72 ----
    const unsigned short* src = XT + (size_t)(bh * 8 + ti) * 196 * 64;
    if (wave == 0) {
        tdm_load_rows64_pad72(src, (unsigned)(uintptr_t)(void*)&Asp[0], 196u);
    }
    __syncthreads();

    // build transposed copy from LDS
    for (int i = tid; i < 196 * 64; i += 256) {
        const int si = i >> 6, dd = i & 63;
        Atp[dd * 232 + si] = Asp[si * 72 + dd];
    }
    __syncthreads();

    // ---- S = scale * A A^T (13x13 tiles of 16x16, K = 64 = 2 WMMA steps) ----
    for (int idx = wave; idx < 169; idx += 8) {
        const int qt = idx / 13, nt = idx % 13;
        v8f acc = v8f_zero();
#pragma unroll
        for (int ks = 0; ks < 2; ++ks) {
            v16bf a  = load_frag_row(Asp + (qt * 16 + lrow) * 72 + ks * 32, halfSel);
            v16bf bb = load_frag_row(Asp + (nt * 16 + lrow) * 72 + ks * 32, halfSel);
            acc = wmma_bf16(a, bb, acc);
        }
#pragma unroll
        for (int j = 0; j < 8; ++j) {
            const int r = qt * 16 + halfSel * 8 + j;
            Sp[r * 232 + nt * 16 + lrow] = f32_to_bf16_bits(acc[j] * 0.125f);
        }
    }
    __syncthreads();

    // ---- row softmax over valid 196 cols; zero the K-pad cols ----
    if (tid < 208) {
        unsigned short* row = Sp + tid * 232;
        float mx = -1e30f;
        for (int c = 0; c < 196; ++c) mx = fmaxf(mx, bf16_bits_to_f32(row[c]));
        float sum = 0.f;
        for (int c = 0; c < 196; ++c) {
            const float e = __expf(bf16_bits_to_f32(row[c]) - mx);
            sum += e;
            row[c] = f32_to_bf16_bits(e);
        }
        const float inv = 1.0f / sum;
        for (int c = 0; c < 196; ++c)
            row[c] = f32_to_bf16_bits(bf16_bits_to_f32(row[c]) * inv);
        for (int c = 196; c < 224; ++c) row[c] = 0;
    }
    __syncthreads();

    // ---- O = P @ A : per wave 16-row q-tiles, K padded to 224 (7 steps) ----
    for (int qt = wave; qt < 13; qt += 8) {
        v8f o[4];
#pragma unroll
        for (int u = 0; u < 4; ++u) o[u] = v8f_zero();
        for (int ks = 0; ks < 7; ++ks) {
            v16bf a = load_frag_row(Sp + (qt * 16 + lrow) * 232 + ks * 32, halfSel);
#pragma unroll
            for (int u = 0; u < 4; ++u) {
                v16bf bb = load_frag_row(Atp + (u * 16 + lrow) * 232 + ks * 32, halfSel);
                o[u] = wmma_bf16(a, bb, o[u]);
            }
        }
#pragma unroll
        for (int u = 0; u < 4; ++u) {
            const int dd = u * 16 + lrow;
#pragma unroll
            for (int j = 0; j < 8; ++j) {
                const int si = qt * 16 + halfSel * 8 + j;
                if (si < 196) {
                    const size_t r = (size_t)((si * 8 + ti) * 16 + b);
                    XO[r * 768 + h * 64 + dd] = f32_to_bf16_bits(o[u][j]);
                }
            }
        }
    }
}

// ---------------------------------------------------------------------------
// Host-side orchestration (workspace layout, ~197 MB, with aliasing)
// ---------------------------------------------------------------------------
static constexpr size_t SZ_W1  = (size_t)2304 * 768 * 2;
static constexpr size_t SZ_W2  = (size_t)768 * 768 * 2;
static constexpr size_t SZ_XB  = (size_t)25088 * 768 * 2;
static constexpr size_t SZ_QKV = (size_t)25088 * 2304 * 2;
static constexpr size_t OFF_W1  = 0;
static constexpr size_t OFF_W2  = OFF_W1 + SZ_W1;
static constexpr size_t OFF_XB  = OFF_W2 + SZ_W2;      // X bf16, reused as XO
static constexpr size_t OFF_QKV = OFF_XB + SZ_XB;
static constexpr size_t OFF_XT  = OFF_QKV + SZ_QKV;

extern "C" void kernel_launch(void* const* d_in, const int* in_sizes, int n_in,
                              void* d_out, int out_size, void* d_ws, size_t ws_size,
                              hipStream_t stream) {
    const float* query = (const float*)d_in[0];   // (1568, 16, 768)
    const float* w_in  = (const float*)d_in[1];   // (2304, 768)
    const float* b_in  = (const float*)d_in[2];   // (2304,)
    const float* w_out = (const float*)d_in[3];   // (768, 768)
    const float* b_out = (const float*)d_in[4];   // (768,)
    float* out = (float*)d_out;                   // (1568, 16, 768) f32

    char* ws = (char*)d_ws;
    unsigned short* W1  = (unsigned short*)(ws + OFF_W1);
    unsigned short* W2  = (unsigned short*)(ws + OFF_W2);
    unsigned short* XB  = (unsigned short*)(ws + OFF_XB);   // also XO
    unsigned short* QKV = (unsigned short*)(ws + OFF_QKV);
    unsigned short* XT  = (unsigned short*)(ws + OFF_XT);

    // converts
    tsa_convert_bf16<<<8192, 256, 0, stream>>>(query, XB, 25088 * 768);
    tsa_convert_bf16<<<4096, 256, 0, stream>>>(w_in,  W1, 2304 * 768);
    tsa_convert_bf16<<<2048, 256, 0, stream>>>(w_out, W2, 768 * 768);

    // QKV = X @ W_in^T + b_in  (bf16 out)
    tsa_gemm_bf16<0><<<dim3(18, 196), 256, 0, stream>>>(
        XB, W1, b_in, nullptr, QKV, 25088, 2304, 768);

    // temporal attention -> XT [bh][t][s][d]
    tsa_temporal_attn<<<4704, 256, 0, stream>>>(QKV, XT);

    // spatial attention -> XO (reuses XB region), layout (L*B, 768)
    tsa_spatial_attn<<<1536, 256, 0, stream>>>(XT, XB);

    // out = XO @ W_out^T + b_out  (f32 out)
    tsa_gemm_bf16<1><<<dim3(6, 196), 256, 0, stream>>>(
        XB, W2, b_out, out, nullptr, 25088, 768, 768);
}
